// AttentionFreeTransformerConv_13365938226135
// MI455X (gfx1250) — matrix operand order repaired and checked by reading, not compiled, not run on hardware
//
#include <hip/hip_runtime.h>
#include <hip/hip_bf16.h>

// Problem constants (match reference)
#define B_    16
#define D_    128
#define L_    8192
#define K_    16
#define LOUT_ (L_ - K_ + 1)   // 8177
#define TN    32              // t-tile width per block (2 WMMA subtiles)
#define DC    32              // din chunk per staging round (= WMMA K dim)
#define TP    48              // staged t positions: TN + K - 1 = 47, pad to 48
#define RS    48              // LDS row stride in halves (96 B: aligned + bank spread)

typedef __attribute__((ext_vector_type(16))) _Float16 v16h;
typedef __attribute__((ext_vector_type(8)))  _Float16 v8h;
typedef __attribute__((ext_vector_type(8)))  float    v8f;

// ---------------------------------------------------------------------------
// Zero the two global scalar accumulators.
__global__ void init_sums(float* sums) {
    if (threadIdx.x == 0) { sums[0] = 0.f; sums[1] = 0.f; }
}

// wtT[k][dout][din] = f16(exp(w[dout][din][k]) - 1), computed once (262144).
__global__ void transform_weights(const float* __restrict__ w,
                                  _Float16* __restrict__ wtT, int n) {
    int i = blockIdx.x * blockDim.x + threadIdx.x;   // input order (dout,din,k)
    if (i < n) {
        const int k    = i & (K_ - 1);
        const int din  = (i >> 4) & (D_ - 1);
        const int dout = i >> 11;
        wtT[((size_t)k * D_ + dout) * D_ + din] = (_Float16)(__expf(w[i]) - 1.0f);
    }
}

// S1 = sum(exp(key)*value), S2 = sum(exp(key)) over all B*D*L elements.
__global__ __launch_bounds__(256)
void reduce_sums(const float* __restrict__ key, const float* __restrict__ value,
                 float* sums, size_t n) {
    __shared__ float r1[256];
    __shared__ float r2[256];
    const int tid = threadIdx.x;
    float s1 = 0.f, s2 = 0.f;
    for (size_t i = (size_t)blockIdx.x * blockDim.x + tid; i < n;
         i += (size_t)gridDim.x * blockDim.x) {
        float e = __expf(key[i]);
        s1 += e * value[i];
        s2 += e;
    }
    r1[tid] = s1; r2[tid] = s2;
    __syncthreads();
    for (int s = 128; s > 0; s >>= 1) {
        if (tid < s) { r1[tid] += r1[tid + s]; r2[tid] += r2[tid + s]; }
        __syncthreads();
    }
    if (tid == 0) {
        atomicAdd(&sums[0], r1[0]);
        atomicAdd(&sums[1], r2[0]);
    }
}

// ---------------------------------------------------------------------------
// Conv as K=16 shifted GEMMs with WMMA f16->f32.
//   out[b,dout,t] = sigmoid(q) * (conv(key*value) + S1) / (conv(key) + S2)
// Block: (batch b, 32-wide t tile); 8 waves, wave w owns dout rows [16w,16w+16).
// Per 32-din chunk: stage X^T (key, key*value) once in LDS — zero-redundancy,
// coalesced — then issue 16 taps x 2 subtiles x 2 convs = 64 WMMAs per round.
__global__ __launch_bounds__(256)
void aft_conv_wmma(const float* __restrict__ query,
                   const float* __restrict__ key,
                   const float* __restrict__ value,
                   const _Float16* __restrict__ wtT,
                   const float* __restrict__ sums,
                   float* __restrict__ out) {
    const int t0    = blockIdx.x * TN;
    const int b     = blockIdx.y;
    const int tid   = threadIdx.x;
    const int wave  = tid >> 5;      // wave32
    const int lane  = tid & 31;
    const int dout0 = wave * 16;
    const int m     = lane & 15;     // A row (dout) / B column (t)
    const int gA    = lane >> 4;     // lane-group: K sub-range select

    // Transposed staged tiles: [t position][din 0..31], f16.
    __shared__ alignas(32) _Float16 sKT[TP][RS];   // key^T
    __shared__ alignas(32) _Float16 sVT[TP][RS];   // (key*value)^T

    v8f cd[2] = {};   // den conv accumulators (2 t subtiles)
    v8f cn[2] = {};   // num conv accumulators

    for (int dc = 0; dc < D_; dc += DC) {
        __syncthreads();   // prior round's ds reads done before overwrite
        // Stage DC x TP elements of key / key*value, transposed. 6 iters/thread.
        for (int idx = tid; idx < DC * TP; idx += 256) {
            const int din = idx / TP;
            const int tp  = idx - din * TP;
            const int t   = t0 + tp;
            const size_t rowb = ((size_t)b * D_ + dc + din) * L_;
            float kx = 0.f, vx = 0.f;
            if (t < L_) {
                kx = key[rowb + t];
                vx = value[rowb + t];
                if (dc + DC < D_)   // speculative prefetch of next chunk's rows
                    __builtin_prefetch(&key[rowb + (size_t)DC * L_ + t], 0, 1);
            }
            sKT[tp][din] = (_Float16)kx;
            sVT[tp][din] = (_Float16)(kx * vx);
        }
        __syncthreads();

        // Base of this wave's A rows in wtT for this din chunk.
        const _Float16* wbase = wtT + (size_t)(dout0 + m) * D_ + dc + gA * 16;
#pragma unroll
        for (int k = 0; k < K_; ++k) {   // 16 shifted GEMM taps
            // A fragment: W_k 16x32 (dout x din). Contiguous K order (matches B):
            //   lanes 0-15: K0..15
            //   lanes16-31: K16..31
            const _Float16* wk = wbase + (size_t)k * (D_ * D_);
            const v8h alo = *(const v8h*)(wk);
            const v8h ahi = *(const v8h*)(wk + 8);
            const v16h a  = __builtin_shufflevector(
                alo, ahi, 0, 1, 2, 3, 4, 5, 6, 7, 8, 9, 10, 11, 12, 13, 14, 15);

#pragma unroll
            for (int s = 0; s < 2; ++s) {   // two 16-wide t subtiles
                // B fragment (32x16): lane group gA holds K rows gA*16+h, all
                // contiguous in the transposed LDS tile; 32-B aligned slots.
                const int row = k + s * 16 + m;
                const v16h bk = *(const v16h*)&sKT[row][gA * 16];
                const v16h bv = *(const v16h*)&sVT[row][gA * 16];
                cd[s] = __builtin_amdgcn_wmma_f32_16x16x32_f16(
                    false, a, false, bk, (short)0, cd[s], false, false);
                cn[s] = __builtin_amdgcn_wmma_f32_16x16x32_f16(
                    false, a, false, bv, (short)0, cn[s], false, false);
            }
        }
    }

    // Epilogue. C layout: lanes 0-15 -> N=lane, M=r; lanes 16-31 -> N=lane-16,
    // M=r+8.
    const float S1 = sums[0];
    const float S2 = sums[1];
    const int gl = lane >> 4;
#pragma unroll
    for (int s = 0; s < 2; ++s) {
        const int t = t0 + s * 16 + (lane & 15);
        if (t < LOUT_) {
#pragma unroll
            for (int r = 0; r < 8; ++r) {
                const int dout = dout0 + r + gl * 8;
                const size_t idx = ((size_t)b * D_ + dout) * LOUT_ + t;
                const float q   = query[idx];
                const float sig = 1.f / (1.f + __expf(-q));
                out[idx] = sig * (cn[s][r] + S1) / (cd[s][r] + S2);
            }
        }
    }
}

// ---------------------------------------------------------------------------
extern "C" void kernel_launch(void* const* d_in, const int* in_sizes, int n_in,
                              void* d_out, int out_size, void* d_ws, size_t ws_size,
                              hipStream_t stream) {
    const float* query   = (const float*)d_in[0];
    const float* key     = (const float*)d_in[1];
    const float* value   = (const float*)d_in[2];
    const float* weights = (const float*)d_in[3];
    float* out = (float*)d_out;

    // Workspace layout: [0..7] two f32 scalar sums; f16 wtT at +64 (512 KB).
    float*    sums = (float*)d_ws;
    _Float16* wtT  = (_Float16*)((char*)d_ws + 64);

    init_sums<<<1, 32, 0, stream>>>(sums);

    const int nw = D_ * D_ * K_;
    transform_weights<<<(nw + 255) / 256, 256, 0, stream>>>(weights, wtT, nw);

    reduce_sums<<<2048, 256, 0, stream>>>(key, value, sums,
                                          (size_t)B_ * D_ * L_);

    dim3 grid((LOUT_ + TN - 1) / TN, B_);   // 256 x 16 blocks, 8 waves each
    aft_conv_wmma<<<grid, 256, 0, stream>>>(query, key, value, wtT, sums, out);
}
